// DeformableParts_4861902979809
// MI455X (gfx1250) — matrix operand order repaired and checked by reading, not compiled, not run on hardware
//
#include <hip/hip_runtime.h>
#include <hip/hip_bf16.h>
#include <math.h>

// ---------------- types ----------------
typedef __bf16 bf16_t;
typedef __attribute__((ext_vector_type(16))) __bf16 v16bf;
typedef __attribute__((ext_vector_type(8)))  float  v8f;

// ---------------- problem constants ----------------
static constexpr int N_   = 2;
static constexpr int C_   = 128;
static constexpr int H_   = 100;
static constexpr int W_   = 152;
static constexpr int NC_  = 80;
static constexpr int HP_  = H_ + 2;     // padded rows (conv halo)
static constexpr int WP_  = 168;        // padded cols: halo + 32-wide tile margin
static constexpr int HW_  = H_ * W_;
static constexpr int XT_  = 5;          // ceil(152/32) x-tiles of 32
static constexpr int OUTC_= 340;        // 80 logits + 4 obs + 256 pos

__host__ __device__ static inline int cdiv_i(int a, int b) { return (a + b - 1) / b; }

// channel swizzle inside each 32-block so the hardware A-fragment K pattern
// ({0..7,16..23} | {8..15,24..31}) maps to one contiguous 16-element load.
__device__ __forceinline__ int kswz(int r) { return (r >= 8 && r < 24) ? (r ^ 24) : r; }

// ============================================================
// pack input x (NCHW f32) -> padded, channel-swizzled NHWC bf16
// ============================================================
__global__ void pack_x_kernel(const float* __restrict__ x, bf16_t* __restrict__ xp) {
  __shared__ bf16_t t[64][C_];
  int bx = blockIdx.x;
  int xc = bx % 3; bx /= 3;
  int y  = bx % H_;
  int n  = bx / H_;
  int x0 = xc * 64;
  for (int k = threadIdx.x; k < 64 * C_; k += blockDim.x) {
    int c = k >> 6, xi = k & 63, xx = x0 + xi;
    float v = (xx < W_) ? x[(((size_t)n * C_ + c) * H_ + y) * W_ + xx] : 0.f;
    t[xi][c] = (bf16_t)v;
  }
  __syncthreads();
  for (int k = threadIdx.x; k < 64 * C_; k += blockDim.x) {
    int xi = k >> 7, c = k & 127, xx = x0 + xi;
    if (xx < W_) {
      int slot = (c & 96) | kswz(c & 31);
      xp[(((size_t)n * HP_ + y + 1) * WP_ + xx + 1) * C_ + slot] = t[xi][c];
    }
  }
}

// ============================================================
// pack conv weights [O][128][3][3] f32 -> bf16 B-fragment layout
// wpk[((tap*4+cig)*Op + o)*32 + cl] = w[o][cig*32+kswz(cl)][tap] ; o>=O -> 0
// ============================================================
__global__ void pack_w_kernel(const float* __restrict__ w, bf16_t* __restrict__ wpk,
                              int O, int Op) {
  size_t total = (size_t)36 * Op * 32;
  size_t i = (size_t)blockIdx.x * blockDim.x + threadIdx.x;
  if (i >= total) return;
  int cl = (int)(i & 31);
  size_t t = i >> 5;
  int o     = (int)(t % Op);
  int chunk = (int)(t / Op);
  int cig = chunk & 3, tap = chunk >> 2;
  float v = 0.f;
  if (o < O) v = w[((size_t)o * C_ + cig * 32 + kswz(cl)) * 9 + tap];
  wpk[i] = (bf16_t)v;
}

// ============================================================
// conv3x3 via implicit GEMM with V_WMMA_F32_16X16X32_BF16
//   One wave owns a 32(M:positions) x 32(N:channels) register tile:
//   2 A-fragments x 2 B-fragments -> 4 WMMAs per 4 loads
//   (2x arithmetic intensity, 4 independent accumulator chains).
//   grid.x = N*H*XT (32-position row tiles), grid.y = Op/32
// mode 0: + bias -> NHWC f32 scratch (tower layer, pre-GN)
// mode 1: + bias -> d_out NCHW logits (ch 0..79; Op padded to 96)
// mode 2: obs = loc +/- exp(scale^2*(conv+bias)) -> d_out ch 80..83 (Op=32)
// ============================================================
__global__ __launch_bounds__(32)
void conv3x3_wmma_kernel(const bf16_t* __restrict__ xp, const bf16_t* __restrict__ wpk,
                         const float* __restrict__ bias, float* __restrict__ out,
                         const float* __restrict__ scale_ptr, int Op, int mode) {
  const int lane = threadIdx.x & 31;
  const int mcol = lane & 15;   // A: spatial row / B,D: output-channel column
  const int kofs = lane & 16;   // K half selected by lane group
  int bx = blockIdx.x;
  const int xt = bx % XT_; bx /= XT_;
  const int y  = bx % H_;
  const int n  = bx / H_;
  const int o0 = blockIdx.y * 32;
  const int xb = xt * 32;

  v8f a00 = {}, a01 = {}, a10 = {}, a11 = {};   // acc[mi][ni]
#pragma unroll
  for (int tap = 0; tap < 9; ++tap) {
    const int dy = tap / 3, dx = tap % 3;
    const bf16_t* ap0 = xp  + (((size_t)n * HP_ + (y + dy)) * WP_ + (xb + mcol + dx)) * C_ + kofs;
    const bf16_t* ap1 = ap0 + (size_t)16 * C_;
    const bf16_t* bp0 = wpk + (((size_t)tap * 4) * Op + (o0 + mcol)) * (size_t)32 + kofs;
    const bf16_t* bp1 = bp0 + 16 * 32;
#pragma unroll
    for (int cig = 0; cig < 4; ++cig) {
      v16bf fa0 = *(const v16bf*)(ap0 + cig * 32);
      v16bf fa1 = *(const v16bf*)(ap1 + cig * 32);
      v16bf fb0 = *(const v16bf*)(bp0 + (size_t)cig * Op * 32);
      v16bf fb1 = *(const v16bf*)(bp1 + (size_t)cig * Op * 32);
      a00 = __builtin_amdgcn_wmma_f32_16x16x32_bf16(false, fa0, false, fb0, (short)0, a00, false, false);
      a01 = __builtin_amdgcn_wmma_f32_16x16x32_bf16(false, fa0, false, fb1, (short)0, a01, false, false);
      a10 = __builtin_amdgcn_wmma_f32_16x16x32_bf16(false, fa1, false, fb0, (short)0, a10, false, false);
      a11 = __builtin_amdgcn_wmma_f32_16x16x32_bf16(false, fa1, false, fb1, (short)0, a11, false, false);
    }
  }

  const int mb = (lane < 16) ? 0 : 8;   // D: VGPR r holds M = r (+8 for hi lanes)
#pragma unroll
  for (int ni = 0; ni < 2; ++ni) {
    const int o = o0 + ni * 16 + mcol;
    if (mode == 0) {
      const float bv = bias[o];
#pragma unroll
      for (int mi = 0; mi < 2; ++mi) {
        const v8f acc = (mi == 0) ? (ni == 0 ? a00 : a01) : (ni == 0 ? a10 : a11);
#pragma unroll
        for (int r = 0; r < 8; ++r) {
          int xx = xb + mi * 16 + mb + r;
          if (xx < W_) out[(((size_t)n * HW_) + (size_t)y * W_ + xx) * C_ + o] = acc[r] + bv;
        }
      }
    } else if (mode == 1) {
      if (o < NC_) {
        const float bv = bias[o];
        size_t base = ((size_t)n * OUTC_ + o) * HW_ + (size_t)y * W_;
#pragma unroll
        for (int mi = 0; mi < 2; ++mi) {
          const v8f acc = (mi == 0) ? (ni == 0 ? a00 : a01) : (ni == 0 ? a10 : a11);
#pragma unroll
          for (int r = 0; r < 8; ++r) {
            int xx = xb + mi * 16 + mb + r;
            if (xx < W_) out[base + xx] = acc[r] + bv;
          }
        }
      }
    } else {
      if (o < 4) {
        const float bv = bias[o];
        const float s  = scale_ptr[0];
        const float s2 = s * s;
        const float ly = (float)(y * 8 + 4);
        size_t base = ((size_t)n * OUTC_ + 80 + o) * HW_ + (size_t)y * W_;
#pragma unroll
        for (int mi = 0; mi < 2; ++mi) {
          const v8f acc = (mi == 0) ? (ni == 0 ? a00 : a01) : (ni == 0 ? a10 : a11);
#pragma unroll
          for (int r = 0; r < 8; ++r) {
            int xx = xb + mi * 16 + mb + r;
            if (xx < W_) {
              float v  = expf(s2 * (acc[r] + bv));
              float lx = (float)(xx * 8 + 4);
              float ob = (o == 0) ? lx - v : (o == 1) ? ly - v : (o == 2) ? lx + v : ly + v;
              out[base + xx] = ob;
            }
          }
        }
      }
    }
  }
}

// ============================================================
// GroupNorm: per-(n, group of 4 channels) mean/rstd over H*W*4
// ============================================================
__global__ void gn_reduce_kernel(const float* __restrict__ buf, float* __restrict__ stats) {
  int n = blockIdx.x >> 5, g = blockIdx.x & 31;
  float s = 0.f, ss = 0.f;
  for (int pos = threadIdx.x; pos < HW_; pos += blockDim.x) {
    const float4 v = *(const float4*)(buf + ((size_t)n * HW_ + pos) * C_ + g * 4);
    s  += v.x + v.y + v.z + v.w;
    ss += v.x * v.x + v.y * v.y + v.z * v.z + v.w * v.w;
  }
  __shared__ float r1[256], r2[256];
  r1[threadIdx.x] = s; r2[threadIdx.x] = ss;
  __syncthreads();
  for (int k = 128; k > 0; k >>= 1) {
    if ((int)threadIdx.x < k) { r1[threadIdx.x] += r1[threadIdx.x + k];
                                r2[threadIdx.x] += r2[threadIdx.x + k]; }
    __syncthreads();
  }
  if (threadIdx.x == 0) {
    const float inv = 1.f / (4.f * HW_);
    float m   = r1[0] * inv;
    float var = r2[0] * inv - m * m;
    stats[((size_t)n * 32 + g) * 2 + 0] = m;
    stats[((size_t)n * 32 + g) * 2 + 1] = rsqrtf(var + 1e-5f);
  }
}

// GN apply + ReLU + bf16 repack into padded swizzled NHWC (next layer input)
__global__ void gn_apply_pack_kernel(const float* __restrict__ buf, const float* __restrict__ stats,
                                     const float* __restrict__ gamma, const float* __restrict__ beta,
                                     bf16_t* __restrict__ xp) {
  size_t i = (size_t)blockIdx.x * blockDim.x + threadIdx.x;
  size_t total = (size_t)N_ * HW_ * C_;
  if (i >= total) return;
  int c = (int)(i & 127);
  size_t p = i >> 7;
  int xx = (int)(p % W_);
  int yy = (int)((p / W_) % H_);
  int n  = (int)(p / ((size_t)W_ * H_));
  int g  = c >> 2;
  float m = stats[((size_t)n * 32 + g) * 2 + 0];
  float r = stats[((size_t)n * 32 + g) * 2 + 1];
  float v = (buf[i] - m) * r * gamma[c] + beta[c];
  v = v > 0.f ? v : 0.f;
  int slot = (c & 96) | kswz(c & 31);
  xp[(((size_t)n * HP_ + yy + 1) * WP_ + xx + 1) * C_ + slot] = (bf16_t)v;
}

// ============================================================
// mask cumsums (general; mask happens to be all-false)
// ============================================================
__global__ void cumsum_y_kernel(const unsigned char* __restrict__ mask, float* __restrict__ ye) {
  int i = blockIdx.x * blockDim.x + threadIdx.x;
  if (i >= N_ * W_) return;
  int n = i / W_, xx = i % W_;
  float acc = 0.f;
  for (int y = 0; y < H_; ++y) {
    acc += 1.f - (mask[((size_t)n * H_ + y) * W_ + xx] ? 1.f : 0.f);
    ye[((size_t)n * H_ + y) * W_ + xx] = acc;
  }
  float sc = 6.283185307179586f / (acc + 1e-6f);
  for (int y = 0; y < H_; ++y) ye[((size_t)n * H_ + y) * W_ + xx] *= sc;
}

__global__ void cumsum_x_kernel(const unsigned char* __restrict__ mask, float* __restrict__ xe) {
  int i = blockIdx.x * blockDim.x + threadIdx.x;
  if (i >= N_ * H_) return;
  int n = i / H_, y = i % H_;
  const unsigned char* mr = mask + ((size_t)n * H_ + y) * W_;
  float* er = xe + ((size_t)n * H_ + y) * W_;
  float acc = 0.f;
  for (int xx = 0; xx < W_; ++xx) { acc += 1.f - (mr[xx] ? 1.f : 0.f); er[xx] = acc; }
  float sc = 6.283185307179586f / (acc + 1e-6f);
  for (int xx = 0; xx < W_; ++xx) er[xx] *= sc;
}

// pos_y (ch 84..147) and pos_x (ch 148..211)
__global__ void pos_xy_kernel(const float* __restrict__ ye, const float* __restrict__ xe,
                              float* __restrict__ out) {
  size_t i = (size_t)blockIdx.x * blockDim.x + threadIdx.x;
  size_t total = (size_t)N_ * 64 * HW_;
  if (i >= total) return;
  int pos = (int)(i % HW_);
  int d   = (int)((i / HW_) % 64);
  int n   = (int)(i / ((size_t)HW_ * 64));
  float dim = __powf(10000.f, (float)(2 * (d >> 1)) / 64.f);
  float ay = ye[(size_t)n * HW_ + pos] / dim;
  float ax = xe[(size_t)n * HW_ + pos] / dim;
  out[((size_t)n * OUTC_ +  84 + d) * HW_ + pos] = (d & 1) ? cosf(ay) : sinf(ay);
  out[((size_t)n * OUTC_ + 148 + d) * HW_ + pos] = (d & 1) ? cosf(ax) : sinf(ax);
}

// pos_c = 1x1 conv(sigmoid(logits)) -> ch 212..275 ; logits staged in LDS per row
__global__ void pos_c_kernel(const float* __restrict__ dout, const float* __restrict__ pw,
                             const float* __restrict__ pb, float* __restrict__ out) {
  __shared__ float sl[NC_][W_ + 1];
  int y = blockIdx.x % H_;
  int n = blockIdx.x / H_;
  for (int k = threadIdx.x; k < NC_ * W_; k += blockDim.x) {
    int c = k / W_, xx = k % W_;
    float v = dout[((size_t)n * OUTC_ + c) * HW_ + (size_t)y * W_ + xx];
    sl[c][xx] = 1.f / (1.f + expf(-v));
  }
  __syncthreads();
  for (int k = threadIdx.x; k < 64 * W_; k += blockDim.x) {
    int d = k / W_, xx = k % W_;
    float s = pb[d];
    for (int c = 0; c < NC_; ++c) s += sl[c][xx] * pw[d * NC_ + c];
    out[((size_t)n * OUTC_ + 212 + d) * HW_ + (size_t)y * W_ + xx] = s;
  }
}

// pos_d = sine embedding of obs boxes -> ch 276..339
__global__ void pos_d_kernel(float* __restrict__ out) {
  size_t i = (size_t)blockIdx.x * blockDim.x + threadIdx.x;
  size_t total = (size_t)N_ * 64 * HW_;
  if (i >= total) return;
  int pos = (int)(i % HW_);
  int q   = (int)((i / HW_) % 64);
  int n   = (int)(i / ((size_t)HW_ * 64));
  int coord = q >> 4, j = q & 15;
  float dim = __powf(10000.f, (float)(2 * (j >> 1)) / 16.f);
  float ob  = out[((size_t)n * OUTC_ + 80 + coord) * HW_ + pos] / dim;
  out[((size_t)n * OUTC_ + 276 + q) * HW_ + pos] = (j & 1) ? cosf(ob) : sinf(ob);
}

// ============================================================
// launcher
// ============================================================
extern "C" void kernel_launch(void* const* d_in, const int* in_sizes, int n_in,
                              void* d_out, int out_size, void* d_ws, size_t ws_size,
                              hipStream_t stream) {
  (void)in_sizes; (void)n_in; (void)out_size; (void)ws_size;
  const float*         x        = (const float*)d_in[0];
  const unsigned char* mask     = (const unsigned char*)d_in[1];
  const float*         cls_w    = (const float*)d_in[2];
  const float*         cls_b    = (const float*)d_in[3];
  const float*         cls_gn_g = (const float*)d_in[4];
  const float*         cls_gn_b = (const float*)d_in[5];
  const float*         box_w    = (const float*)d_in[6];
  const float*         box_b    = (const float*)d_in[7];
  const float*         box_gn_g = (const float*)d_in[8];
  const float*         box_gn_b = (const float*)d_in[9];
  const float*         logits_w = (const float*)d_in[10];
  const float*         logits_b = (const float*)d_in[11];
  const float*         boxes_w  = (const float*)d_in[12];
  const float*         boxes_b  = (const float*)d_in[13];
  const float*         scale    = (const float*)d_in[14];
  const float*         proj_w   = (const float*)d_in[15];
  const float*         proj_b   = (const float*)d_in[16];
  float* out = (float*)d_out;
  char*  ws  = (char*)d_ws;

  // ---- workspace layout (256B aligned) ----
  size_t off = 0;
  auto take = [&](size_t bytes) { size_t o = off; off = (off + bytes + 255) & ~(size_t)255; return o; };
  const size_t XP_BYTES = (size_t)N_ * HP_ * WP_ * C_ * sizeof(bf16_t);  // 8.77 MB
  size_t xp0_o  = take(XP_BYTES);
  size_t xp1_o  = take(XP_BYTES);
  size_t xp2_o  = take(XP_BYTES);
  size_t conv_o = take((size_t)N_ * HW_ * C_ * sizeof(float));           // 15.6 MB
  size_t stat_o = take((size_t)N_ * 32 * 2 * sizeof(float));
  size_t ye_o   = take((size_t)N_ * HW_ * sizeof(float));
  size_t xe_o   = take((size_t)N_ * HW_ * sizeof(float));
  size_t wc0_o  = take((size_t)36 * 128 * 32 * sizeof(bf16_t));
  size_t wc1_o  = take((size_t)36 * 128 * 32 * sizeof(bf16_t));
  size_t wb0_o  = take((size_t)36 * 128 * 32 * sizeof(bf16_t));
  size_t wb1_o  = take((size_t)36 * 128 * 32 * sizeof(bf16_t));
  size_t wlg_o  = take((size_t)36 * 96  * 32 * sizeof(bf16_t));          // logits padded 80->96
  size_t wbx_o  = take((size_t)36 * 32  * 32 * sizeof(bf16_t));          // boxes  padded  4->32

  bf16_t* xp0 = (bf16_t*)(ws + xp0_o);
  bf16_t* xp1 = (bf16_t*)(ws + xp1_o);
  bf16_t* xp2 = (bf16_t*)(ws + xp2_o);
  float*  cv  = (float*)(ws + conv_o);
  float*  st  = (float*)(ws + stat_o);
  float*  ye  = (float*)(ws + ye_o);
  float*  xe  = (float*)(ws + xe_o);
  bf16_t* wc0 = (bf16_t*)(ws + wc0_o);
  bf16_t* wc1 = (bf16_t*)(ws + wc1_o);
  bf16_t* wb0 = (bf16_t*)(ws + wb0_o);
  bf16_t* wb1 = (bf16_t*)(ws + wb1_o);
  bf16_t* wlg = (bf16_t*)(ws + wlg_o);
  bf16_t* wbx = (bf16_t*)(ws + wbx_o);

  // zero padded activation buffers (pads must be 0 every launch)
  hipMemsetAsync(xp0, 0, 3 * XP_BYTES, stream);

  // stage input + pack weights (bf16 fragment layouts)
  pack_x_kernel<<<N_ * H_ * 3, 256, 0, stream>>>(x, xp0);
  const size_t WSTRIDE = (size_t)C_ * C_ * 9;
  pack_w_kernel<<<cdiv_i(36 * 128 * 32, 256), 256, 0, stream>>>(cls_w,            wc0, 128, 128);
  pack_w_kernel<<<cdiv_i(36 * 128 * 32, 256), 256, 0, stream>>>(cls_w + WSTRIDE,  wc1, 128, 128);
  pack_w_kernel<<<cdiv_i(36 * 128 * 32, 256), 256, 0, stream>>>(box_w,            wb0, 128, 128);
  pack_w_kernel<<<cdiv_i(36 * 128 * 32, 256), 256, 0, stream>>>(box_w + WSTRIDE,  wb1, 128, 128);
  pack_w_kernel<<<cdiv_i(36 * 96 * 32, 256),  256, 0, stream>>>(logits_w,         wlg, 80, 96);
  pack_w_kernel<<<cdiv_i(36 * 32 * 32, 256),  256, 0, stream>>>(boxes_w,          wbx, 4, 32);

  const dim3 gT(N_ * H_ * XT_, 4), gL(N_ * H_ * XT_, 3), gB(N_ * H_ * XT_, 1);
  const int gGN = cdiv_i(N_ * HW_ * C_, 256);

  // ---- cls tower ----
  conv3x3_wmma_kernel<<<gT, 32, 0, stream>>>(xp0, wc0, cls_b,       cv, scale, 128, 0);
  gn_reduce_kernel<<<N_ * 32, 256, 0, stream>>>(cv, st);
  gn_apply_pack_kernel<<<gGN, 256, 0, stream>>>(cv, st, cls_gn_g,        cls_gn_b,        xp1);
  conv3x3_wmma_kernel<<<gT, 32, 0, stream>>>(xp1, wc1, cls_b + 128, cv, scale, 128, 0);
  gn_reduce_kernel<<<N_ * 32, 256, 0, stream>>>(cv, st);
  gn_apply_pack_kernel<<<gGN, 256, 0, stream>>>(cv, st, cls_gn_g + 128,  cls_gn_b + 128,  xp2);
  conv3x3_wmma_kernel<<<gL, 32, 0, stream>>>(xp2, wlg, logits_b,    out, scale, 96, 1);

  // ---- box tower ----
  conv3x3_wmma_kernel<<<gT, 32, 0, stream>>>(xp0, wb0, box_b,       cv, scale, 128, 0);
  gn_reduce_kernel<<<N_ * 32, 256, 0, stream>>>(cv, st);
  gn_apply_pack_kernel<<<gGN, 256, 0, stream>>>(cv, st, box_gn_g,        box_gn_b,        xp1);
  conv3x3_wmma_kernel<<<gT, 32, 0, stream>>>(xp1, wb1, box_b + 128, cv, scale, 128, 0);
  gn_reduce_kernel<<<N_ * 32, 256, 0, stream>>>(cv, st);
  gn_apply_pack_kernel<<<gGN, 256, 0, stream>>>(cv, st, box_gn_g + 128,  box_gn_b + 128,  xp2);
  conv3x3_wmma_kernel<<<gB, 32, 0, stream>>>(xp2, wbx, boxes_b,     out, scale, 32, 2);

  // ---- positional embeddings ----
  cumsum_y_kernel<<<cdiv_i(N_ * W_, 128), 128, 0, stream>>>(mask, ye);
  cumsum_x_kernel<<<cdiv_i(N_ * H_, 128), 128, 0, stream>>>(mask, xe);
  int posTot = (int)((size_t)N_ * 64 * HW_);
  pos_xy_kernel<<<cdiv_i(posTot, 256), 256, 0, stream>>>(ye, xe, out);
  pos_c_kernel<<<N_ * H_, 256, 0, stream>>>(out, proj_w, proj_b, out);
  pos_d_kernel<<<cdiv_i(posTot, 256), 256, 0, stream>>>(out);
}
